// ECGCNN_MoE_Small_1005022347835
// MI455X (gfx1250) — compile-verified, matrix operand order checked by reading, not run on hardware
//
#include <hip/hip_runtime.h>
#include <hip/hip_bf16.h>
#include <cstdint>

// CDNA5 / gfx1250: wave32, WMMA 16x16x32 f16 -> f32, TDM async tensor loads.
typedef __attribute__((ext_vector_type(16))) _Float16 v16h;
typedef __attribute__((ext_vector_type(8)))  float    v8f;

// K-slot mapping shared by A and B fragments:
// lanes 0-15 hold k_local = j (j=0..15), lanes 16-31 hold k_local = 16+j.
__device__ __forceinline__ int klocal16(int lane, int j) {
    return (lane < 16) ? j : (16 + j);
}

// ---------------------------------------------------------------------------
// Tensor Data Mover: 2D tile DMA global -> LDS (CDNA5 ISA/§8).
// Guarded so the file still builds (direct-load fallback) on toolchains
// without the builtin. Arity differs: 5 args (ROCm7.2/clang-22), 6 (clang-23).
// ---------------------------------------------------------------------------
#if __has_builtin(__builtin_amdgcn_tensor_load_to_lds) && __has_builtin(__builtin_amdgcn_s_wait_tensorcnt)
#define HAVE_TDM 1
#else
#define HAVE_TDM 0
#endif

#if HAVE_TDM
typedef __attribute__((ext_vector_type(4))) unsigned int u32x4;
typedef __attribute__((ext_vector_type(8))) int          i32x8;
typedef __attribute__((ext_vector_type(4))) int          i32x4;

// Load a [tile_y rows x tile_x cols] fp32 tile from a row-major tensor with
// row stride `stride_x` elements into contiguous LDS at `lds_dst`.
__device__ __forceinline__ void tdm_load_2d(void* lds_dst, const float* gsrc,
                                            unsigned tile_x, unsigned tile_y,
                                            unsigned tensor_x, unsigned tensor_y,
                                            unsigned stride_x) {
    unsigned long long ga = (unsigned long long)(uintptr_t)gsrc;
    // Generic LDS pointer: addr[31:0] is the LDS byte offset (aperture layout).
    unsigned lds_off = (unsigned)(uintptr_t)lds_dst;

    u32x4 g0;
    g0[0] = 1u;                                    // count=1, user mode
    g0[1] = lds_off;                               // D#.lds_addr (bits 63:32)
    g0[2] = (unsigned)(ga & 0xffffffffu);          // global_addr[31:0]
    g0[3] = (unsigned)((ga >> 32) & 0x01ffffffu)   // global_addr[56:32]
          | (2u << 30);                            // type=2 ("image")

    i32x8 g1;
    g1[0] = (int)(2u << 16);                       // data_size=2 (4 bytes), mask=0
    g1[1] = (int)((tensor_x & 0xffffu) << 16);     // tensor_dim0[15:0] @ bits 63:48
    g1[2] = (int)((tensor_x >> 16) | ((tensor_y & 0xffffu) << 16));
    g1[3] = (int)(((tensor_y >> 16) & 0xffffu) | (tile_x << 16));   // tile_dim0
    g1[4] = (int)(tile_y & 0xffffu);               // tile_dim1 (tile_dim2 = 0)
    g1[5] = (int)stride_x;                         // tensor_dim0_stride[31:0]
    g1[6] = 0;                                     // stride0 hi / stride1 lo
    g1[7] = 0;

    i32x4 z4 = {0, 0, 0, 0};
#if __clang_major__ >= 23
    i32x8 z8 = {0, 0, 0, 0, 0, 0, 0, 0};
    __builtin_amdgcn_tensor_load_to_lds(g0, g1, z4, z4, z8, 0);
#else
    __builtin_amdgcn_tensor_load_to_lds(g0, g1, z4, z4, 0);
#endif
}
#endif  // HAVE_TDM

// ---------------------------------------------------------------------------
// Stem: conv1d(1->8,k3,same) + bias + ReLU, plus mean over L for the router.
// ---------------------------------------------------------------------------
__global__ __launch_bounds__(256)
void conv1_pool_kernel(const float* __restrict__ x, const float* __restrict__ w,
                       const float* __restrict__ bias, float* __restrict__ h,
                       float* __restrict__ pooled) {
    const int b = blockIdx.x;
    const int tid = threadIdx.x;
    const int L = 2048;
    __shared__ float red[256 * 8];

    float wreg[24];
#pragma unroll
    for (int i = 0; i < 24; ++i) wreg[i] = w[i];
    float breg[8];
#pragma unroll
    for (int i = 0; i < 8; ++i) breg[i] = bias[i];

    const float* xb = x + (size_t)b * L;
    float sum[8];
#pragma unroll
    for (int co = 0; co < 8; ++co) sum[co] = 0.f;

    for (int p = 0; p < 8; ++p) {
        int l = tid + p * 256;
        float xm = (l > 0)     ? xb[l - 1] : 0.f;
        float x0 = xb[l];
        float xp = (l < L - 1) ? xb[l + 1] : 0.f;
#pragma unroll
        for (int co = 0; co < 8; ++co) {
            float y = wreg[co * 3 + 0] * xm + wreg[co * 3 + 1] * x0 +
                      wreg[co * 3 + 2] * xp + breg[co];
            y = fmaxf(y, 0.f);
            h[((size_t)b * 8 + co) * L + l] = y;
            sum[co] += y;
        }
    }
#pragma unroll
    for (int co = 0; co < 8; ++co) red[tid * 8 + co] = sum[co];
    __syncthreads();
    for (int s = 128; s > 0; s >>= 1) {
        if (tid < s)
            for (int co = 0; co < 8; ++co)
                red[tid * 8 + co] += red[(tid + s) * 8 + co];
        __syncthreads();
    }
    if (tid < 8) pooled[b * 8 + tid] = red[tid] * (1.0f / 2048.0f);
}

// ---------------------------------------------------------------------------
// Router: logits -> softmax -> top-2 renormalized dense gates + probs sums.
// ---------------------------------------------------------------------------
__global__ __launch_bounds__(256)
void router_kernel(const float* __restrict__ pooled, const float* __restrict__ rw,
                   const float* __restrict__ rb, float* __restrict__ gate,
                   float* __restrict__ probs_sum, int B) {
    int b = blockIdx.x * blockDim.x + threadIdx.x;
    if (b >= B) return;
    float p8[8];
#pragma unroll
    for (int k = 0; k < 8; ++k) p8[k] = pooled[b * 8 + k];
    float lg[4];
#pragma unroll
    for (int e = 0; e < 4; ++e) {
        float s = rb[e];
#pragma unroll
        for (int k = 0; k < 8; ++k) s += p8[k] * rw[e * 8 + k];
        lg[e] = s;
    }
    float mx = fmaxf(fmaxf(lg[0], lg[1]), fmaxf(lg[2], lg[3]));
    float pe[4];
    float se = 0.f;
#pragma unroll
    for (int e = 0; e < 4; ++e) { pe[e] = __expf(lg[e] - mx); se += pe[e]; }
#pragma unroll
    for (int e = 0; e < 4; ++e) {
        pe[e] /= se;
        atomicAdd(&probs_sum[e], pe[e]);
    }
    int i1 = 0;
    for (int e = 1; e < 4; ++e) if (pe[e] > pe[i1]) i1 = e;
    int i2 = -1;
    for (int e = 0; e < 4; ++e) {
        if (e == i1) continue;
        if (i2 < 0 || pe[e] > pe[i2]) i2 = e;
    }
    float gs = pe[i1] + pe[i2];
    for (int e = 0; e < 4; ++e)
        gate[b * 4 + e] = (e == i1) ? pe[i1] / gs : ((e == i2) ? pe[i2] / gs : 0.f);
}

__global__ void aux_kernel(const float* __restrict__ probs_sum,
                           float* __restrict__ out_aux, int B) {
    if (threadIdx.x != 0 || blockIdx.x != 0) return;
    float mp[4], mean = 0.f;
    for (int e = 0; e < 4; ++e) { mp[e] = probs_sum[e] / (float)B; mean += mp[e]; }
    mean *= 0.25f;
    float var = 0.f;
    for (int e = 0; e < 4; ++e) { float d = mp[e] - mean; var += d * d; }
    var /= 3.0f;  // ddof=1
    float cv = sqrtf(var) / (mean + 1e-10f);
    *out_aux = cv * cv;
}

__global__ void zero_kernel(float* p, int n) {
    int i = blockIdx.x * blockDim.x + threadIdx.x;
    if (i < n) p[i] = 0.f;
}

// ---------------------------------------------------------------------------
// Expert stage (blocks 1 & 2): conv k3 same as implicit GEMM via WMMA,
// fused bias+BN+ReLU+maxpool2. Input tile [CIN,128] TDM-staged into LDS;
// +/-1 halo columns in separate LDS slots (edge zeros folded at fill time).
// Grid: (LIN/128, B, E); uniform skip of gate==0 experts.
// ---------------------------------------------------------------------------
template <int CIN, int COUT, int LIN, bool IN_E>
__global__ __launch_bounds__(256)
void expert_stage_kernel(const float* __restrict__ in_base,
                         const float* __restrict__ ew,   // [E,COUT,CIN,3]
                         const float* __restrict__ eb,   // [E,COUT]
                         const float* __restrict__ bng, const float* __restrict__ bnb,
                         const float* __restrict__ bnm, const float* __restrict__ bnv,
                         float* __restrict__ out_base,   // [E,B,COUT,LIN/2]
                         const float* __restrict__ gate, int B) {
    constexpr int KTOT   = 3 * CIN;
    constexpr int KSTEPS = (KTOT + 31) / 32;
    constexpr int KPAD   = KSTEPS * 32;
    constexpr int MT     = COUT / 16;
    constexpr int LOUT   = LIN / 2;

    __shared__ _Float16 wlds[COUT * KPAD];
    __shared__ float tile[CIN * 128];
    __shared__ float haloL[CIN];
    __shared__ float haloR[CIN];
    __shared__ float bnsc[COUT];
    __shared__ float bnsh[COUT];

    const int e = blockIdx.z;
    const int b = blockIdx.y;
    const int chunk = blockIdx.x;

    // MoE sparsity: gate is exactly 0 for non-top-2 experts; uniform branch.
    if (gate[b * 4 + e] == 0.f) return;

    const int base = chunk * 128;
    const size_t samp = IN_E ? ((size_t)e * B + b) : (size_t)b;
    const float* inp = in_base + samp * CIN * LIN;

#if HAVE_TDM
    // One wave issues the tile DMA; overlaps with weight/BN LDS fill below.
    if ((threadIdx.x >> 5) == 0)
        tdm_load_2d(tile, inp + base, 128u, (unsigned)CIN,
                    (unsigned)LIN, (unsigned)CIN, (unsigned)LIN);
#else
    for (int idx = threadIdx.x; idx < CIN * 128; idx += 256) {
        int ci = idx >> 7, p = idx & 127;
        tile[idx] = inp[(size_t)ci * LIN + base + p];
    }
#endif

    const float* W = ew + (size_t)e * COUT * CIN * 3;
    for (int idx = threadIdx.x; idx < COUT * KPAD; idx += 256) {
        int co = idx / KPAD, k = idx - co * KPAD;
        float wv = 0.f;
        if (k < KTOT) {
            int ci = k / 3, dk = k - ci * 3;
            wv = W[(co * CIN + ci) * 3 + dk];
        }
        wlds[idx] = (_Float16)wv;
    }
    for (int co = threadIdx.x; co < COUT; co += 256) {
        float sc = bng[e * COUT + co] * rsqrtf(bnv[e * COUT + co] + 1e-5f);
        bnsc[co] = sc;
        bnsh[co] = (eb[e * COUT + co] - bnm[e * COUT + co]) * sc + bnb[e * COUT + co];
    }
    for (int ci = threadIdx.x; ci < CIN; ci += 256) {
        haloL[ci] = (base > 0)         ? inp[(size_t)ci * LIN + base - 1]   : 0.f;
        haloR[ci] = (base + 128 < LIN) ? inp[(size_t)ci * LIN + base + 128] : 0.f;
    }

#if HAVE_TDM
    if ((threadIdx.x >> 5) == 0) __builtin_amdgcn_s_wait_tensorcnt(0);
#endif
    __syncthreads();

    const int lane = threadIdx.x & 31;
    const int wv = threadIdx.x >> 5;
    const int n0 = base + wv * 16;

    const v8f zero8 = {0.f, 0.f, 0.f, 0.f, 0.f, 0.f, 0.f, 0.f};
    v8f acc[MT];
#pragma unroll
    for (int t = 0; t < MT; ++t) acc[t] = zero8;

#pragma unroll
    for (int ks = 0; ks < KSTEPS; ++ks) {
        v16h bf;  // im2col fragment from LDS tile
#pragma unroll
        for (int j = 0; j < 16; ++j) {
            int k = ks * 32 + klocal16(lane, j);
            float val = 0.f;
            if (k < KTOT) {
                int ci = k / 3, dk = k - ci * 3;
                int p = wv * 16 + (lane & 15) + dk - 1;  // local col in tile
                val = (p < 0) ? haloL[ci]
                              : ((p >= 128) ? haloR[ci] : tile[ci * 128 + p]);
            }
            bf[j] = (_Float16)val;
        }
#pragma unroll
        for (int t = 0; t < MT; ++t) {
            v16h af;
#pragma unroll
            for (int j = 0; j < 16; ++j) {
                int k = ks * 32 + klocal16(lane, j);
                af[j] = wlds[(t * 16 + (lane & 15)) * KPAD + k];
            }
            acc[t] = __builtin_amdgcn_wmma_f32_16x16x32_f16(
                false, af, false, bf, (short)0, acc[t], false, false);
        }
    }

    // Epilogue: BN + ReLU + maxpool2 via lane-pair shuffle.
    const int mbase = (lane < 16) ? 0 : 8;
    float* outp = out_base + ((size_t)e * B + b) * COUT * LOUT;
#pragma unroll
    for (int t = 0; t < MT; ++t) {
#pragma unroll
        for (int r = 0; r < 8; ++r) {
            int mrow = t * 16 + mbase + r;
            float y = acc[t][r] * bnsc[mrow] + bnsh[mrow];
            y = fmaxf(y, 0.f);
            float pl = fmaxf(y, __shfl_xor(y, 1, 32));
            if ((lane & 1) == 0) {
                int pout = (n0 >> 1) + ((lane & 15) >> 1);
                outp[(size_t)mrow * LOUT + pout] = pl;
            }
        }
    }
}

// ---------------------------------------------------------------------------
// Stage 3 + gate-weighted combine: loops over experts in-register, skipping
// gate==0 experts (uniform). Per-expert weight tile + TDM input tile in LDS.
// Writes combined [B,64,256] directly.
// ---------------------------------------------------------------------------
__global__ __launch_bounds__(256)
void stage3_combine_kernel(const float* __restrict__ s2,   // [E,B,32,512]
                           const float* __restrict__ ew,   // [E,64,32,3]
                           const float* __restrict__ eb,
                           const float* __restrict__ bng, const float* __restrict__ bnb,
                           const float* __restrict__ bnm, const float* __restrict__ bnv,
                           const float* __restrict__ gate,
                           float* __restrict__ combined,   // [B,64,256]
                           int B) {
    constexpr int CIN = 32, COUT = 64, LIN = 512;
    constexpr int KTOT = 96, KSTEPS = 3, KPAD = 96, MT = 4, LOUT = 256, E = 4;

    __shared__ _Float16 wlds[COUT * KPAD];  // 12 KB, reloaded per gated expert
    __shared__ float tile[CIN * 128];       // 16 KB TDM tile
    __shared__ float haloL[CIN];
    __shared__ float haloR[CIN];
    __shared__ float bnsc[E * COUT];
    __shared__ float bnsh[E * COUT];

    const int b = blockIdx.y;
    const int chunk = blockIdx.x;
    const int base = chunk * 128;

    for (int ec = threadIdx.x; ec < E * COUT; ec += 256) {
        float sc = bng[ec] * rsqrtf(bnv[ec] + 1e-5f);
        bnsc[ec] = sc;
        bnsh[ec] = (eb[ec] - bnm[ec]) * sc + bnb[ec];
    }

    const int lane = threadIdx.x & 31;
    const int wv = threadIdx.x >> 5;
    const int n0 = base + wv * 16;
    const int mbase = (lane < 16) ? 0 : 8;
    const v8f zero8 = {0.f, 0.f, 0.f, 0.f, 0.f, 0.f, 0.f, 0.f};

    float res[MT][8];
#pragma unroll
    for (int t = 0; t < MT; ++t)
#pragma unroll
        for (int r = 0; r < 8; ++r) res[t][r] = 0.f;

    for (int e = 0; e < E; ++e) {
        float ge = gate[b * 4 + e];
        if (ge == 0.f) continue;  // uniform: only top-2 experts computed
        const float* inp = s2 + ((size_t)e * B + b) * CIN * LIN;

        __syncthreads();  // previous expert's LDS reads complete
#if HAVE_TDM
        if ((threadIdx.x >> 5) == 0)
            tdm_load_2d(tile, inp + base, 128u, (unsigned)CIN,
                        (unsigned)LIN, (unsigned)CIN, (unsigned)LIN);
#else
        for (int idx = threadIdx.x; idx < CIN * 128; idx += 256) {
            int ci = idx >> 7, p = idx & 127;
            tile[idx] = inp[(size_t)ci * LIN + base + p];
        }
#endif
        for (int idx = threadIdx.x; idx < COUT * KPAD; idx += 256) {
            int co = idx / KPAD, k = idx - co * KPAD;  // KTOT == KPAD
            int ci = k / 3, dk = k - ci * 3;
            wlds[idx] = (_Float16)ew[((size_t)e * COUT + co) * CIN * 3 + ci * 3 + dk];
        }
        for (int ci = threadIdx.x; ci < CIN; ci += 256) {
            haloL[ci] = (base > 0)         ? inp[(size_t)ci * LIN + base - 1]   : 0.f;
            haloR[ci] = (base + 128 < LIN) ? inp[(size_t)ci * LIN + base + 128] : 0.f;
        }
#if HAVE_TDM
        if ((threadIdx.x >> 5) == 0) __builtin_amdgcn_s_wait_tensorcnt(0);
#endif
        __syncthreads();

        v8f acc[MT];
#pragma unroll
        for (int t = 0; t < MT; ++t) acc[t] = zero8;

#pragma unroll
        for (int ks = 0; ks < KSTEPS; ++ks) {
            v16h bf;
#pragma unroll
            for (int j = 0; j < 16; ++j) {
                int k = ks * 32 + klocal16(lane, j);
                int ci = k / 3, dk = k - ci * 3;
                int p = wv * 16 + (lane & 15) + dk - 1;
                float val = (p < 0) ? haloL[ci]
                                    : ((p >= 128) ? haloR[ci] : tile[ci * 128 + p]);
                bf[j] = (_Float16)val;
            }
#pragma unroll
            for (int t = 0; t < MT; ++t) {
                v16h af;
#pragma unroll
                for (int j = 0; j < 16; ++j) {
                    int k = ks * 32 + klocal16(lane, j);
                    af[j] = wlds[(t * 16 + (lane & 15)) * KPAD + k];
                }
                acc[t] = __builtin_amdgcn_wmma_f32_16x16x32_f16(
                    false, af, false, bf, (short)0, acc[t], false, false);
            }
        }

#pragma unroll
        for (int t = 0; t < MT; ++t) {
#pragma unroll
            for (int r = 0; r < 8; ++r) {
                int mrow = t * 16 + mbase + r;
                float y = acc[t][r] * bnsc[e * COUT + mrow] + bnsh[e * COUT + mrow];
                y = fmaxf(y, 0.f);
                float pl = fmaxf(y, __shfl_xor(y, 1, 32));
                res[t][r] += ge * pl;  // gate-weighted combine, fused
            }
        }
    }

    float* outp = combined + (size_t)b * COUT * LOUT;
#pragma unroll
    for (int t = 0; t < MT; ++t) {
#pragma unroll
        for (int r = 0; r < 8; ++r) {
            if ((lane & 1) == 0) {
                int mrow = t * 16 + mbase + r;
                int pout = (n0 >> 1) + ((lane & 15) >> 1);
                outp[(size_t)mrow * LOUT + pout] = res[t][r];
            }
        }
    }
}

// ---------------------------------------------------------------------------
// fc1: [B,16384] x [16384,64]^T + bias, ReLU. WMMA GEMM, M=batch, N=64.
// ---------------------------------------------------------------------------
__global__ __launch_bounds__(128)
void fc1_kernel(const float* __restrict__ combined, const float* __restrict__ fc1w,
                const float* __restrict__ fc1b, float* __restrict__ z, int B) {
    const int lane = threadIdx.x & 31;
    const int wv = threadIdx.x >> 5;  // 0..3
    const int m0 = blockIdx.x * 16;
    const int n0 = wv * 16;
    const int K = 16384;
    const int mrowA = m0 + (lane & 15);
    const int nrow = n0 + (lane & 15);
    const float* arow = combined + (size_t)mrowA * K;
    const float* brow = fc1w + (size_t)nrow * K;

    v8f acc = {0.f, 0.f, 0.f, 0.f, 0.f, 0.f, 0.f, 0.f};
    for (int ks = 0; ks < K / 32; ++ks) {
        int kb = ks * 32 + ((lane < 16) ? 0 : 16);
        __builtin_prefetch(arow + kb + 256, 0, 1);  // global_prefetch_b8
        __builtin_prefetch(brow + kb + 256, 0, 1);
        v16h af, bf;
#pragma unroll
        for (int j = 0; j < 16; ++j) {
            af[j] = (_Float16)arow[kb + j];
            bf[j] = (_Float16)brow[kb + j];
        }
        acc = __builtin_amdgcn_wmma_f32_16x16x32_f16(
            false, af, false, bf, (short)0, acc, false, false);
    }
    const int mbase = (lane < 16) ? 0 : 8;
#pragma unroll
    for (int r = 0; r < 8; ++r) {
        int m = m0 + mbase + r;
        int n = n0 + (lane & 15);
        z[(size_t)m * 64 + n] = fmaxf(acc[r] + fc1b[n], 0.f);
    }
}

__global__ __launch_bounds__(256)
void fc2_kernel(const float* __restrict__ z, const float* __restrict__ w2,
                const float* __restrict__ b2, float* __restrict__ out, int B) {
    int b = blockIdx.x * blockDim.x + threadIdx.x;
    if (b >= B) return;
    float zr[64];
#pragma unroll
    for (int k = 0; k < 64; ++k) zr[k] = z[(size_t)b * 64 + k];
#pragma unroll
    for (int c = 0; c < 5; ++c) {
        float s = b2[c];
#pragma unroll
        for (int k = 0; k < 64; ++k) s += zr[k] * w2[c * 64 + k];
        out[(size_t)b * 5 + c] = s;
    }
}

// ---------------------------------------------------------------------------
extern "C" void kernel_launch(void* const* d_in, const int* in_sizes, int n_in,
                              void* d_out, int out_size, void* d_ws, size_t ws_size,
                              hipStream_t stream) {
    (void)in_sizes; (void)n_in; (void)out_size; (void)ws_size;
    const float* x    = (const float*)d_in[0];
    const float* c1w  = (const float*)d_in[1];
    const float* c1b  = (const float*)d_in[2];
    const float* ew1  = (const float*)d_in[3];
    const float* eb1  = (const float*)d_in[4];
    const float* g1   = (const float*)d_in[5];
    const float* bb1  = (const float*)d_in[6];
    const float* m1   = (const float*)d_in[7];
    const float* v1   = (const float*)d_in[8];
    const float* ew2  = (const float*)d_in[9];
    const float* eb2  = (const float*)d_in[10];
    const float* g2   = (const float*)d_in[11];
    const float* bb2  = (const float*)d_in[12];
    const float* m2   = (const float*)d_in[13];
    const float* v2   = (const float*)d_in[14];
    const float* ew3  = (const float*)d_in[15];
    const float* eb3  = (const float*)d_in[16];
    const float* g3   = (const float*)d_in[17];
    const float* bb3  = (const float*)d_in[18];
    const float* m3   = (const float*)d_in[19];
    const float* v3   = (const float*)d_in[20];
    const float* rw   = (const float*)d_in[21];
    const float* rb   = (const float*)d_in[22];
    const float* fc1w = (const float*)d_in[23];
    const float* fc1b = (const float*)d_in[24];
    const float* fc2w = (const float*)d_in[25];
    const float* fc2b = (const float*)d_in[26];
    float* out = (float*)d_out;

    const int B = 1024, E = 4;
    char* ws = (char*)d_ws;
    size_t off = 0;
    auto alloc = [&](size_t bytes) -> float* {
        float* p = (float*)(ws + off);
        off += (bytes + 255) & ~(size_t)255;
        return p;
    };
    float* h      = alloc((size_t)B * 8 * 2048 * 4);        // stem activations
    float* s1     = alloc((size_t)E * B * 16 * 1024 * 4);   // stage1 out
    float* s2     = alloc((size_t)E * B * 32 * 512 * 4);    // stage2 out
    float* pooled = alloc((size_t)B * 8 * 4);
    float* gate   = alloc((size_t)B * 4 * 4);
    float* zbuf   = alloc((size_t)B * 64 * 4);
    float* psum   = alloc(16);
    float* combined = s1;  // s1 is dead after stage2 reads it; reuse its space

    zero_kernel<<<1, 32, 0, stream>>>(psum, 4);
    conv1_pool_kernel<<<B, 256, 0, stream>>>(x, c1w, c1b, h, pooled);
    router_kernel<<<B / 256, 256, 0, stream>>>(pooled, rw, rb, gate, psum, B);
    aux_kernel<<<1, 32, 0, stream>>>(psum, out + (size_t)B * 5, B);

    expert_stage_kernel<8, 16, 2048, false>
        <<<dim3(16, B, E), 256, 0, stream>>>(h, ew1, eb1, g1, bb1, m1, v1, s1, gate, B);
    expert_stage_kernel<16, 32, 1024, true>
        <<<dim3(8, B, E), 256, 0, stream>>>(s1, ew2, eb2, g2, bb2, m2, v2, s2, gate, B);
    stage3_combine_kernel<<<dim3(4, B), 256, 0, stream>>>(
        s2, ew3, eb3, g3, bb3, m3, v3, gate, combined, B);

    fc1_kernel<<<B / 16, 128, 0, stream>>>(combined, fc1w, fc1b, zbuf, B);
    fc2_kernel<<<B / 256, 256, 0, stream>>>(zbuf, fc2w, fc2b, out, B);
}
